// GAT_1116691497134
// MI455X (gfx1250) — matrix operand order repaired and checked by reading, not compiled, not run on hardware
//
#include <hip/hip_runtime.h>
#include <hip/hip_bf16.h>

// ---- problem dims (match reference) ----
#define B_  8
#define N_  1024
#define F_  256
#define O_  128
#define H_  8
#define NEGV (-2.147483648e9f)   // -2^31 torch padding value
#define ALPHA_ 0.2f
#define LSTR 40                  // LDS tile row stride (elems): 80B, 16B-aligned rows

typedef __attribute__((ext_vector_type(16))) __bf16    v16bf;
typedef __attribute__((ext_vector_type(8)))  float     v8f;
typedef __attribute__((ext_vector_type(4)))  unsigned  v4u;
typedef __attribute__((ext_vector_type(8)))  int       v8i;
typedef __attribute__((ext_vector_type(4)))  int       v4i;

// ---------------- gfx1250 data-mover feature gates ----------------
#if __has_builtin(__builtin_amdgcn_global_load_async_to_lds_b128)
#define HAVE_ASYNC_LDS 1
#warning "CDNA5 probe: global_load_async_to_lds_b128 path ENABLED"
#else
#warning "CDNA5 probe: global_load_async_to_lds_b128 path DISABLED (sync fallback)"
#endif
#if __has_builtin(__builtin_amdgcn_tensor_load_to_lds)
#define HAVE_TDM 1
#warning "CDNA5 probe: tensor_load_to_lds (TDM) path ENABLED"
#else
#warning "CDNA5 probe: tensor_load_to_lds (TDM) path DISABLED (fallback staging)"
#endif
#if __has_builtin(__builtin_amdgcn_s_wait_asynccnt)
#warning "CDNA5 probe: s_wait_asynccnt builtin ENABLED"
#endif
#if __has_builtin(__builtin_amdgcn_s_wait_tensorcnt)
#warning "CDNA5 probe: s_wait_tensorcnt builtin ENABLED"
#endif

#if __has_builtin(__builtin_amdgcn_s_wait_asynccnt)
#define WAIT_ASYNCCNT(n) __builtin_amdgcn_s_wait_asynccnt(n)
#elif defined(HAVE_ASYNC_LDS)
#define WAIT_ASYNCCNT(n) asm volatile("s_wait_asynccnt %0" :: "n"(n) : "memory")
#else
#define WAIT_ASYNCCNT(n) do {} while (0)
#endif

#if __has_builtin(__builtin_amdgcn_s_wait_tensorcnt)
#define WAIT_TENSORCNT(n) __builtin_amdgcn_s_wait_tensorcnt((short)(n))
#elif defined(HAVE_TDM)
#define WAIT_TENSORCNT(n) asm volatile("s_wait_tensorcnt %0" :: "n"(n) : "memory")
#else
#define WAIT_TENSORCNT(n) do {} while (0)
#endif

static __device__ __forceinline__ float leaky_(float x) {
    return x > 0.f ? x : ALPHA_ * x;
}

// ---- WMMA bf16 16x16x32, f32 accumulate ----
#define WMMA_BF16(ACC, AF, BF)                                                  \
    (ACC) = __builtin_amdgcn_wmma_f32_16x16x32_bf16(false, (AF), false, (BF),   \
                                                    (short)0, (ACC), false, false)

// A fragment (16 rows x 32 K) from LDS row-major tile (stride LSTR).
static __device__ __forceinline__ v16bf load_a_frag(const __bf16* aS, int rbase, int lane) {
    const int hf = (lane >> 4) & 1, m = lane & 15;
    union { v16bf v; unsigned u[8]; } f;
    const __bf16* row = aS + (rbase + m) * LSTR;
#pragma unroll
    for (int v = 0; v < 8; ++v) {
        const int k = (v < 4 ? 2 * v : 2 * v + 8) + hf * 8;
        f.u[v] = *(const unsigned*)(row + k);
    }
    return f.v;
}

// B fragment (32 K x 16 cols) from LDS tile stored [col][k] (stride LSTR).
static __device__ __forceinline__ v16bf load_b_frag(const __bf16* wS, int cbase, int lane) {
    const int hf = (lane >> 4) & 1, n = lane & 15;
    union { v16bf v; unsigned u[8]; } f;
    const __bf16* col = wS + (cbase + n) * LSTR + hf * 8;
#pragma unroll
    for (int v = 0; v < 8; ++v) f.u[v] = *(const unsigned*)(col + (v < 4 ? 2 * v : 2 * v + 8));
    return f.v;
}

// ---------------- async / TDM staging helpers ----------------
// Copy one 16B chunk global->LDS (async on gfx1250 path, sync fallback).
// Toolchain declares the builtin with pointer-to-int-vector_size(16) params.
static __device__ __forceinline__ void async_cp16(const void* g, void* l) {
#if defined(HAVE_ASYNC_LDS)
    typedef int vi4 __attribute__((vector_size(16)));
    __builtin_amdgcn_global_load_async_to_lds_b128(
        (vi4*)(uintptr_t)g, (vi4*)(uintptr_t)l, 0, 0);
#else
    *(uint4*)l = *(const uint4*)g;
#endif
}

// Stage a 128row x 32col bf16 tile, direct layout copy (row stride gs elems -> LSTR).
// 512 x 16B chunks over 256 threads => 2 async ops in flight per thread per tile.
static __device__ __forceinline__ void stage_tile_direct(const __bf16* __restrict__ g,
                                                         int gs, __bf16* l, int tid) {
#pragma unroll
    for (int c = 0; c < 2; ++c) {
        const int idx = tid + c * 256;
        const int row = idx >> 2, ko = (idx & 3) * 8;
        async_cp16(g + (size_t)row * gs + ko, l + row * LSTR + ko);
    }
}

#if defined(HAVE_TDM)
// TDM: one descriptor moves a 16row x 32col bf16 tile (2D), hardware-padded to
// LSTR (row 64B = 16 DW, pad 16B = 4 DW). Issued once per wave, TENSORcnt-tracked.
static __device__ __forceinline__ void tdm_stage_rows16(const __bf16* gsrc, int gs_elems,
                                                        __bf16* ldsdst) {
    const unsigned long long ga = (unsigned long long)(uintptr_t)gsrc;
    const v4u g0 = { 1u,                                      // count=1, user mode
                     (unsigned)(uintptr_t)(void*)ldsdst,      // lds_addr
                     (unsigned)ga,                            // global_addr lo
                     (unsigned)((ga >> 32) & 0x01FFFFFFull) | (2u << 30) }; // hi | type=2
    const v8i g1 = { (int)((1u << 16)      // data_size = 2B
                   | (1u << 20)            // pad_enable
                   | (3u << 22)            // pad_interval: 16 DWORDs
                   | (3u << 25)),          // pad_amount:   4 DWORDs
                     (int)(32u << 16),     // tensor_dim0 = 32
                     (int)(16u << 16),     // tensor_dim1 = 16
                     (int)(32u << 16),     // tile_dim0 = 32
                     16,                   // tile_dim1 = 16
                     gs_elems,             // tensor_dim0_stride
                     0, 0 };
    const v4i z4 = { 0, 0, 0, 0 };
#if __clang_major__ >= 23
    const v8i z8 = { 0, 0, 0, 0, 0, 0, 0, 0 };
    __builtin_amdgcn_tensor_load_to_lds(g0, g1, z4, z4, z8, 0);
#else
    __builtin_amdgcn_tensor_load_to_lds(g0, g1, z4, z4, 0);
#endif
}
#define STAGE_B_TILE(gbase, gs, ldsbase, tid, wv)                                     \
    tdm_stage_rows16((gbase) + (size_t)(wv) * 16 * (gs), (gs), (ldsbase) + (wv) * 16 * LSTR)
#define WAIT_B(n) WAIT_TENSORCNT(n)
#else
#define STAGE_B_TILE(gbase, gs, ldsbase, tid, wv)                                     \
    stage_tile_direct((gbase), (gs), (ldsbase), (tid))
#define WAIT_B(n) WAIT_ASYNCCNT((n) * 2)
#endif

// ============================================================================
// K0a/K0b: pre-transpose weights to bf16 [out][k] so GEMM B tiles are direct copies.
// ============================================================================
__global__ __launch_bounds__(256) void k_twmh(const float* __restrict__ W, __bf16* __restrict__ Wmt) {
    const int gid = blockIdx.x * 256 + threadIdx.x;           // (h, o, f)
    const int f = gid & (F_ - 1), o = (gid >> 8) & (O_ - 1), h = gid >> 15;
    Wmt[gid] = (__bf16)W[(size_t)(h * F_ + f) * O_ + o];
}
__global__ __launch_bounds__(256) void k_twout(const float* __restrict__ W, __bf16* __restrict__ Wot) {
    const int gid = blockIdx.x * 256 + threadIdx.x;           // (o, k)
    const int k = gid & (H_ * O_ - 1), o = gid >> 10;
    Wot[gid] = (__bf16)W[(size_t)k * O_ + o];
}

// ============================================================================
// K1: Wh^T[b,h,o,i] from X[b] (NxF) @ W_mh[h]; A staged sync (f32->bf16 cvt),
// B = Wmt rows via async-to-LDS, double-buffered ping-pong.
// ============================================================================
__global__ __launch_bounds__(256) void k_gemm_wh(const float* __restrict__ X,
                                                 const __bf16* __restrict__ Wmt,
                                                 __bf16* __restrict__ Wht) {
    const int b = blockIdx.z, h = blockIdx.y, i0 = blockIdx.x * 128;
    const int tid = threadIdx.x, lane = tid & 31, wv = tid >> 5;
    __shared__ __align__(16) __bf16 aS[2][128 * LSTR];
    __shared__ __align__(16) __bf16 bS[2][128 * LSTR];
    const float* Xb = X + (size_t)b * N_ * F_;
    const __bf16* Wm = Wmt + (size_t)h * O_ * F_;
    v8f acc[8] = {};

    stage_tile_direct(Wm, F_, bS[0], tid);                    // B(0) in flight
#pragma unroll
    for (int r = 0; r < 16; ++r) {                            // A(0) sync cvt
        const int idx = tid + r * 256, m = idx >> 5, k = idx & 31;
        aS[0][m * LSTR + k] = (__bf16)Xb[(size_t)(i0 + m) * F_ + k];
    }
    for (int kc = 0; kc < F_ / 32; ++kc) {
        const int cur = kc & 1, nxt = cur ^ 1;
        if (kc + 1 < F_ / 32) {
            stage_tile_direct(Wm + (kc + 1) * 32, F_, bS[nxt], tid);
#pragma unroll
            for (int r = 0; r < 16; ++r) {
                const int idx = tid + r * 256, m = idx >> 5, k = idx & 31;
                aS[nxt][m * LSTR + k] = (__bf16)Xb[(size_t)(i0 + m) * F_ + (kc + 1) * 32 + k];
            }
            WAIT_ASYNCCNT(2);                                 // tile kc landed, kc+1 in flight
        } else {
            WAIT_ASYNCCNT(0);
        }
        __syncthreads();
        const v16bf af = load_a_frag(aS[cur], wv * 16, lane);
#pragma unroll
        for (int nt = 0; nt < 8; ++nt) {
            const v16bf bf = load_b_frag(bS[cur], nt * 16, lane);
            WMMA_BF16(acc[nt], af, bf);
        }
        __syncthreads();
    }
    // transposed store: Wht[(bh*O+o)*N + i], 8 consecutive i per lane -> 16B packed
    __bf16* outb = Wht + (size_t)(b * H_ + h) * O_ * N_;
    const int hf = lane >> 4, ln = lane & 15;
    const int rowb = i0 + wv * 16 + hf * 8;
#pragma unroll
    for (int nt = 0; nt < 8; ++nt) {
        union { __bf16 hv[8]; uint4 q; } pk;
#pragma unroll
        for (int r = 0; r < 8; ++r) pk.hv[r] = (__bf16)acc[nt][r];
        *(uint4*)(outb + (size_t)(nt * 16 + ln) * N_ + rowb) = pk.q;
    }
}

// ============================================================================
// K2: e1/e2 from transposed Wh: loop over o is coalesced over i across lanes.
// ============================================================================
__global__ __launch_bounds__(256) void k_e1e2(const __bf16* __restrict__ Wht,
                                              const float* __restrict__ a_mh,
                                              float* __restrict__ e1t,
                                              float* __restrict__ e2t) {
    const int gid = blockIdx.x * 256 + threadIdx.x;           // bh*N + i
    const int i = gid & (N_ - 1), bh = gid >> 10, h = bh & (H_ - 1), b = bh >> 3;
    const __bf16* col = Wht + (size_t)bh * O_ * N_ + i;
    const float* a1 = a_mh + h * 2 * O_;
    const float* a2 = a1 + O_;
    float s1 = 0.f, s2 = 0.f;
#pragma unroll 8
    for (int o = 0; o < O_; ++o) {
        const float w = (float)col[(size_t)o * N_];
        s1 += w * a1[o]; s2 += w * a2[o];
    }
    e1t[(size_t)(b * N_ + i) * H_ + h] = s1;
    e2t[(size_t)(b * N_ + i) * H_ + h] = s2;
}

// ============================================================================
// K3: layer-1 column softmax stats (online max/sum over i), all 8 heads at once.
// ============================================================================
__global__ __launch_bounds__(256) void k_colstats1(const float* __restrict__ e1t,
                                                   const float* __restrict__ e2t,
                                                   const int* __restrict__ adj,
                                                   float* __restrict__ cm,
                                                   float* __restrict__ cs) {
    const int b = blockIdx.y, j = blockIdx.x * 256 + threadIdx.x;
    float m[H_], s[H_], e2v[H_];
#pragma unroll
    for (int h = 0; h < H_; ++h) {
        m[h] = -3.0e38f; s[h] = 0.f;
        e2v[h] = e2t[(size_t)(b * N_ + j) * H_ + h];
    }
    const int* adjb = adj + (size_t)b * N_ * N_;
    for (int i = 0; i < N_; ++i) {
        if (i + 8 < N_) __builtin_prefetch(adjb + (size_t)(i + 8) * N_ + j, 0, 1);
        const int av = adjb[(size_t)i * N_ + j];
        const float* e1row = e1t + (size_t)(b * N_ + i) * H_;
#pragma unroll
        for (int h = 0; h < H_; ++h) {
            const float e = av > 0 ? leaky_(e1row[h] + e2v[h]) : NEGV;
            const float mn = fmaxf(m[h], e);
            s[h] = s[h] * __expf(m[h] - mn) + __expf(e - mn);
            m[h] = mn;
        }
    }
#pragma unroll
    for (int h = 0; h < H_; ++h) {
        cm[(size_t)(b * H_ + h) * N_ + j] = m[h];
        cs[(size_t)(b * H_ + h) * N_ + j] = s[h];
    }
}

// att tile builders (128 x 32 -> LDS bf16, stride LSTR)
static __device__ __forceinline__ void stage_att1(__bf16* dst, const float* e1s,
                                                  const float* e2t, const int* adjb,
                                                  const float* cmh, const float* csh,
                                                  int b, int h, int i0, int j0, int tid) {
#pragma unroll
    for (int r = 0; r < 16; ++r) {
        const int idx = tid + r * 256, il = idx >> 5, jl = idx & 31;
        const int i = i0 + il, j = j0 + jl;
        const int av = adjb[(size_t)i * N_ + j];
        const float e2v = e2t[(size_t)(b * N_ + j) * H_ + h];
        const float e = av > 0 ? leaky_(e1s[il] + e2v) : NEGV;
        dst[il * LSTR + jl] = (__bf16)(__expf(e - cmh[j]) / csh[j]);
    }
}
static __device__ __forceinline__ void stage_att2(__bf16* dst, const float* e1s,
                                                  const float* f2, const int* adjb,
                                                  const float* cm2, const float* cs2,
                                                  int b, int i0, int j0, int tid) {
#pragma unroll
    for (int r = 0; r < 16; ++r) {
        const int idx = tid + r * 256, il = idx >> 5, jl = idx & 31;
        const int i = i0 + il, j = j0 + jl;
        const int av = adjb[(size_t)i * N_ + j];
        const float e = av > 0 ? leaky_(e1s[il] + f2[b * N_ + j]) : NEGV;
        dst[il * LSTR + jl] = (__bf16)(__expf(e - cm2[b * N_ + j]) / cs2[b * N_ + j]);
    }
}

// ============================================================================
// K4: fused layer-1 attention GEMM (flash-style). att regenerated per tile in
// registers -> LDS; Wh^T B tiles staged by TDM (per-wave 16-row descriptor),
// double-buffered under TENSORcnt. Output: x2[b,i,h*O+o] with ELU, bf16.
// ============================================================================
__global__ __launch_bounds__(256) void k_attmm1(const float* __restrict__ e1t,
                                                const float* __restrict__ e2t,
                                                const int* __restrict__ adj,
                                                const float* __restrict__ cm,
                                                const float* __restrict__ cs,
                                                const __bf16* __restrict__ Wht,
                                                __bf16* __restrict__ x2) {
    const int b = blockIdx.z, h = blockIdx.y, i0 = blockIdx.x * 128;
    const int tid = threadIdx.x, lane = tid & 31, wv = tid >> 5;
    __shared__ __align__(16) __bf16 attS[2][128 * LSTR];
    __shared__ __align__(16) __bf16 bS[2][128 * LSTR];
    __shared__ float e1s[128];
    if (tid < 128) e1s[tid] = e1t[(size_t)(b * N_ + i0 + tid) * H_ + h];
    const __bf16* Wsrc = Wht + (size_t)(b * H_ + h) * O_ * N_;   // rows o, stride N
    const int* adjb = adj + (size_t)b * N_ * N_;
    const float* cmh = cm + (size_t)(b * H_ + h) * N_;
    const float* csh = cs + (size_t)(b * H_ + h) * N_;
    v8f acc[8] = {};
    __syncthreads();
    STAGE_B_TILE(Wsrc, N_, bS[0], tid, wv);
    stage_att1(attS[0], e1s, e2t, adjb, cmh, csh, b, h, i0, 0, tid);
    for (int t = 0; t < N_ / 32; ++t) {
        const int cur = t & 1, nxt = cur ^ 1, j0 = t * 32;
        if (t + 1 < N_ / 32) {
            STAGE_B_TILE(Wsrc + j0 + 32, N_, bS[nxt], tid, wv);
            stage_att1(attS[nxt], e1s, e2t, adjb, cmh, csh, b, h, i0, j0 + 32, tid);
            WAIT_B(1);
        } else {
            WAIT_B(0);
        }
        __syncthreads();
        const v16bf af = load_a_frag(attS[cur], wv * 16, lane);
#pragma unroll
        for (int nt = 0; nt < 8; ++nt) {
            const v16bf bf = load_b_frag(bS[cur], nt * 16, lane);
            WMMA_BF16(acc[nt], af, bf);
        }
        __syncthreads();
    }
    __bf16* out = x2 + ((size_t)b * N_ + i0) * (H_ * O_) + h * O_;
    const int hf = lane >> 4, ln = lane & 15;
#pragma unroll
    for (int nt = 0; nt < 8; ++nt)
#pragma unroll
        for (int r = 0; r < 8; ++r) {
            const int row = wv * 16 + hf * 8 + r, col = nt * 16 + ln;
            float v = acc[nt][r];
            v = v > 0.f ? v : __expf(v) - 1.f;                // ELU
            out[(size_t)row * (H_ * O_) + col] = (__bf16)v;
        }
}

// ============================================================================
// K5: Wh2^T[b,o,i] = W_out^T (128x1024) x x2^T: both A (Wot) and B (x2 rows)
// tiles are direct async-to-LDS copies, double-buffered. Also emits Wh2 f32
// row-major for the e-score kernel.
// ============================================================================
__global__ __launch_bounds__(256) void k_gemm2(const __bf16* __restrict__ x2,
                                               const __bf16* __restrict__ Wot,
                                               float* __restrict__ Wh2f,
                                               __bf16* __restrict__ Wh2t) {
    const int b = blockIdx.y, i0 = blockIdx.x * 128;          // output cols = node idx
    const int tid = threadIdx.x, lane = tid & 31, wv = tid >> 5;
    __shared__ __align__(16) __bf16 aS[2][128 * LSTR];
    __shared__ __align__(16) __bf16 bS[2][128 * LSTR];
    const __bf16* xb = x2 + ((size_t)b * N_ + i0) * (H_ * O_);
    v8f acc[8] = {};
    stage_tile_direct(Wot, H_ * O_, aS[0], tid);
    stage_tile_direct(xb, H_ * O_, bS[0], tid);
    for (int kc = 0; kc < (H_ * O_) / 32; ++kc) {
        const int cur = kc & 1, nxt = cur ^ 1;
        if (kc + 1 < (H_ * O_) / 32) {
            stage_tile_direct(Wot + (kc + 1) * 32, H_ * O_, aS[nxt], tid);
            stage_tile_direct(xb + (kc + 1) * 32, H_ * O_, bS[nxt], tid);
            WAIT_ASYNCCNT(4);
        } else {
            WAIT_ASYNCCNT(0);
        }
        __syncthreads();
        const v16bf af = load_a_frag(aS[cur], wv * 16, lane); // rows = o
#pragma unroll
        for (int nt = 0; nt < 8; ++nt) {
            const v16bf bf = load_b_frag(bS[cur], nt * 16, lane); // cols = i
            WMMA_BF16(acc[nt], af, bf);
        }
        __syncthreads();
    }
    const int hf = lane >> 4, ln = lane & 15;
#pragma unroll
    for (int nt = 0; nt < 8; ++nt) {
        const int ig = i0 + nt * 16 + ln;
        union { float f[8]; uint4 q[2]; } pf;
#pragma unroll
        for (int r = 0; r < 8; ++r) {
            const float v = acc[nt][r];
            pf.f[r] = v;
            const int o = wv * 16 + hf * 8 + r;
            Wh2t[((size_t)b * O_ + o) * N_ + ig] = (__bf16)v;   // coalesced across lanes
        }
        float* dst = Wh2f + ((size_t)b * N_ + ig) * O_ + wv * 16 + hf * 8;
        *(uint4*)dst = pf.q[0];
        *(uint4*)(dst + 4) = pf.q[1];
    }
}

// ============================================================================
// K6a: f1/f2 row scores. K6b: layer-2 column softmax stats.
// ============================================================================
__global__ __launch_bounds__(256) void k_f1f2(const float* __restrict__ Wh2f,
                                              const float* __restrict__ a_out,
                                              float* __restrict__ f1,
                                              float* __restrict__ f2) {
    const int gid = blockIdx.x * 256 + threadIdx.x;           // b*N + i
    const float* row = Wh2f + (size_t)gid * O_;
    float s1 = 0.f, s2 = 0.f;
#pragma unroll 8
    for (int c = 0; c < O_; ++c) {
        s1 += row[c] * a_out[c];
        s2 += row[c] * a_out[O_ + c];
    }
    f1[gid] = s1; f2[gid] = s2;
}

__global__ __launch_bounds__(256) void k_colstats2(const float* __restrict__ f1,
                                                   const float* __restrict__ f2,
                                                   const int* __restrict__ adj,
                                                   float* __restrict__ cm2,
                                                   float* __restrict__ cs2) {
    const int b = blockIdx.y, j = blockIdx.x * 256 + threadIdx.x;
    float m = -3.0e38f, s = 0.f;
    const float f2v = f2[b * N_ + j];
    const int* adjb = adj + (size_t)b * N_ * N_;
    for (int i = 0; i < N_; ++i) {
        if (i + 8 < N_) __builtin_prefetch(adjb + (size_t)(i + 8) * N_ + j, 0, 1);
        const int av = adjb[(size_t)i * N_ + j];
        const float e = av > 0 ? leaky_(f1[b * N_ + i] + f2v) : NEGV;
        const float mn = fmaxf(m, e);
        s = s * __expf(m - mn) + __expf(e - mn);
        m = mn;
    }
    cm2[b * N_ + j] = m; cs2[b * N_ + j] = s;
}

// ============================================================================
// K7: fused layer-2 attention GEMM + fused output matvecs (h_prime stays in
// accumulators; w_act/w_con dots reduced by __shfl_xor per 16-lane half).
// B tiles = Wh2^T rows via TDM, double-buffered.
// ============================================================================
__global__ __launch_bounds__(256) void k_attmm2(const float* __restrict__ f1,
                                                const float* __restrict__ f2,
                                                const int* __restrict__ adj,
                                                const float* __restrict__ cm2,
                                                const float* __restrict__ cs2,
                                                const __bf16* __restrict__ Wh2t,
                                                const float* __restrict__ w_act,
                                                const float* __restrict__ b_act,
                                                const float* __restrict__ w_con,
                                                const float* __restrict__ b_con,
                                                float* __restrict__ out) {
    const int b = blockIdx.y, i0 = blockIdx.x * 128;
    const int tid = threadIdx.x, lane = tid & 31, wv = tid >> 5;
    __shared__ __align__(16) __bf16 attS[2][128 * LSTR];
    __shared__ __align__(16) __bf16 bS[2][128 * LSTR];
    __shared__ float e1s[128];
    if (tid < 128) e1s[tid] = f1[b * N_ + i0 + tid];
    const __bf16* Wsrc = Wh2t + (size_t)b * O_ * N_;
    const int* adjb = adj + (size_t)b * N_ * N_;
    v8f acc[8] = {};
    __syncthreads();
    STAGE_B_TILE(Wsrc, N_, bS[0], tid, wv);
    stage_att2(attS[0], e1s, f2, adjb, cm2, cs2, b, i0, 0, tid);
    for (int t = 0; t < N_ / 32; ++t) {
        const int cur = t & 1, nxt = cur ^ 1, j0 = t * 32;
        if (t + 1 < N_ / 32) {
            STAGE_B_TILE(Wsrc + j0 + 32, N_, bS[nxt], tid, wv);
            stage_att2(attS[nxt], e1s, f2, adjb, cm2, cs2, b, i0, j0 + 32, tid);
            WAIT_B(1);
        } else {
            WAIT_B(0);
        }
        __syncthreads();
        const v16bf af = load_a_frag(attS[cur], wv * 16, lane);
#pragma unroll
        for (int nt = 0; nt < 8; ++nt) {
            const v16bf bf = load_b_frag(bS[cur], nt * 16, lane);
            WMMA_BF16(acc[nt], af, bf);
        }
        __syncthreads();
    }
    const int hf = lane >> 4, ln = lane & 15;
    float wa[8], wc[8];
#pragma unroll
    for (int nt = 0; nt < 8; ++nt) { wa[nt] = w_act[nt * 16 + ln]; wc[nt] = w_con[nt * 16 + ln]; }
    const float ba = b_act[0], bc = b_con[0];
#pragma unroll
    for (int r = 0; r < 8; ++r) {
        float pa = 0.f, pc = 0.f;
#pragma unroll
        for (int nt = 0; nt < 8; ++nt) {
            const float v = acc[nt][r];
            pa += v * wa[nt]; pc += v * wc[nt];
        }
#pragma unroll
        for (int off = 1; off < 16; off <<= 1) {
            pa += __shfl_xor(pa, off, 32);
            pc += __shfl_xor(pc, off, 32);
        }
        const int row = i0 + wv * 16 + hf * 8 + r;
        if (ln == 0) {
            out[b * N_ + row] = pa + ba;                      // act_pre
            out[B_ * N_ + b * N_ + row] = pc + bc;            // con_pre
        }
    }
}

// ============================================================================
extern "C" void kernel_launch(void* const* d_in, const int* in_sizes, int n_in,
                              void* d_out, int out_size, void* d_ws, size_t ws_size,
                              hipStream_t stream) {
    (void)in_sizes; (void)n_in; (void)out_size; (void)ws_size;
    const float* x_feature = (const float*)d_in[1];   // d_in[0] x_date unused
    const int*   adj       = (const int*)d_in[2];
    const float* W_mh      = (const float*)d_in[3];
    const float* a_mh      = (const float*)d_in[4];
    const float* W_out     = (const float*)d_in[5];
    const float* a_out     = (const float*)d_in[6];
    const float* w_act     = (const float*)d_in[7];
    const float* b_act     = (const float*)d_in[8];
    const float* w_con     = (const float*)d_in[9];
    const float* b_con     = (const float*)d_in[10];
    float* out = (float*)d_out;

    char* wsp = (char*)d_ws;
    auto take = [&](size_t bytes) -> void* {
        void* p = (void*)wsp;
        wsp += (bytes + 255) & ~(size_t)255;
        return p;
    };
    __bf16* Wmt  = (__bf16*)take((size_t)H_ * O_ * F_ * 2);             // 512 KB
    __bf16* Wot  = (__bf16*)take((size_t)O_ * H_ * O_ * 2);             // 256 KB
    __bf16* Wht  = (__bf16*)take((size_t)B_ * H_ * O_ * N_ * 2);        //  16 MB
    __bf16* x2   = (__bf16*)take((size_t)B_ * N_ * H_ * O_ * 2);        //  16 MB
    float*  Wh2f = (float*) take((size_t)B_ * N_ * O_ * 4);             //   4 MB
    __bf16* Wh2t = (__bf16*)take((size_t)B_ * O_ * N_ * 2);             //   2 MB
    float*  e1t  = (float*) take((size_t)B_ * N_ * H_ * 4);
    float*  e2t  = (float*) take((size_t)B_ * N_ * H_ * 4);
    float*  cm1  = (float*) take((size_t)B_ * H_ * N_ * 4);
    float*  cs1  = (float*) take((size_t)B_ * H_ * N_ * 4);
    float*  f1   = (float*) take((size_t)B_ * N_ * 4);
    float*  f2   = (float*) take((size_t)B_ * N_ * 4);
    float*  cm2  = (float*) take((size_t)B_ * N_ * 4);
    float*  cs2  = (float*) take((size_t)B_ * N_ * 4);

    k_twmh     <<<dim3((H_ * O_ * F_) / 256), 256, 0, stream>>>(W_mh, Wmt);
    k_twout    <<<dim3((O_ * H_ * O_) / 256), 256, 0, stream>>>(W_out, Wot);
    k_gemm_wh  <<<dim3(N_ / 128, H_, B_), 256, 0, stream>>>(x_feature, Wmt, Wht);
    k_e1e2     <<<dim3((B_ * H_ * N_) / 256), 256, 0, stream>>>(Wht, a_mh, e1t, e2t);
    k_colstats1<<<dim3(N_ / 256, B_), 256, 0, stream>>>(e1t, e2t, adj, cm1, cs1);
    k_attmm1   <<<dim3(N_ / 128, H_, B_), 256, 0, stream>>>(e1t, e2t, adj, cm1, cs1, Wht, x2);
    k_gemm2    <<<dim3(N_ / 128, B_), 256, 0, stream>>>(x2, Wot, Wh2f, Wh2t);
    k_f1f2     <<<dim3((B_ * N_) / 256), 256, 0, stream>>>(Wh2f, a_out, f1, f2);
    k_colstats2<<<dim3(N_ / 256, B_), 256, 0, stream>>>(f1, f2, adj, cm2, cs2);
    k_attmm2   <<<dim3(N_ / 128, B_), 256, 0, stream>>>(f1, f2, adj, cm2, cs2, Wh2t,
                                                        w_act, b_act, w_con, b_con, out);
}
